// RefQuantLinear_12343736008789
// MI455X (gfx1250) — 8-bit matrix operand order repaired and checked by reading, compiled, not run on hardware
//
#include <hip/hip_runtime.h>
#include <hip/hip_bf16.h>

typedef __attribute__((ext_vector_type(8))) int v8i;
typedef int v4i __attribute__((vector_size(16)));

#define TILE_M 128
#define TILE_N 128
#define TILE_K 64

#define GAS __attribute__((address_space(1)))
#define LAS __attribute__((address_space(3)))

#if __has_builtin(__builtin_amdgcn_global_load_async_to_lds_b128)
#define HAVE_ASYNC_LDS 1
#else
#define HAVE_ASYNC_LDS 0
#endif

__device__ __forceinline__ void async_wait_4() {
#if HAVE_ASYNC_LDS
  __builtin_amdgcn_s_wait_asynccnt(4);
#endif
}
__device__ __forceinline__ void async_wait_0() {
#if HAVE_ASYNC_LDS
  __builtin_amdgcn_s_wait_asynccnt(0);
#endif
}

__global__ __launch_bounds__(256) void quant_input_i8_kernel(
    const float* __restrict__ x, const float* __restrict__ scale,
    unsigned* __restrict__ q4, int n4) {
  int i = blockIdx.x * blockDim.x + threadIdx.x;
  if (i >= n4) return;
  float s = scale[0];
  float4 v = reinterpret_cast<const float4*>(x)[i];
  int b0 = (int)fminf(fmaxf(v.x / s, -128.0f), 127.0f);
  int b1 = (int)fminf(fmaxf(v.y / s, -128.0f), 127.0f);
  int b2 = (int)fminf(fmaxf(v.z / s, -128.0f), 127.0f);
  int b3 = (int)fminf(fmaxf(v.w / s, -128.0f), 127.0f);
  q4[i] = (unsigned)(b0 & 255) | ((unsigned)(b1 & 255) << 8) |
          ((unsigned)(b2 & 255) << 16) | ((unsigned)(b3 & 255) << 24);
}

__global__ __launch_bounds__(256) void quant_weight_i8_kernel(
    const float* __restrict__ w, unsigned* __restrict__ q4, int n4) {
  int i = blockIdx.x * blockDim.x + threadIdx.x;
  if (i >= n4) return;
  float4 v = reinterpret_cast<const float4*>(w)[i];
  int b0 = (int)v.x, b1 = (int)v.y, b2 = (int)v.z, b3 = (int)v.w;
  q4[i] = (unsigned)(b0 & 255) | ((unsigned)(b1 & 255) << 8) |
          ((unsigned)(b2 & 255) << 16) | ((unsigned)(b3 & 255) << 24);
}

__device__ __forceinline__ void stage_tiles(
    const char* __restrict__ Aq, const char* __restrict__ Bq,
    char* __restrict__ As, char* __restrict__ Bs,
    int bm, int bn, int kb, int K, int tid) {
#pragma unroll
  for (int u = 0; u < 2; ++u) {
    int c  = tid + u * 256;
    int r  = c >> 2;
    int cc = (c & 3) << 4;
    const char* ag = Aq + (size_t)(bm + r) * K + kb + cc;
    const char* bg = Bq + (size_t)(bn + r) * K + kb + cc;
    char* al = As + r * 64 + cc;
    char* bl = Bs + r * 64 + cc;
#if HAVE_ASYNC_LDS
    __builtin_amdgcn_global_load_async_to_lds_b128(
        (GAS v4i*)ag, (LAS v4i*)al, 0, 0);
    __builtin_amdgcn_global_load_async_to_lds_b128(
        (GAS v4i*)bg, (LAS v4i*)bl, 0, 0);
#else
    *(uint4*)al = *(const uint4*)ag;
    *(uint4*)bl = *(const uint4*)bg;
#endif
  }
}

__device__ __forceinline__ void compute_tile(
    const char* __restrict__ As, const char* __restrict__ Bs,
    v8i acc[4][2], int wm, int wn, int lrow, int lhalf) {
  v8i bfrag[2];
#pragma unroll
  for (int nt = 0; nt < 2; ++nt) {
    const char* bp = Bs + (wn * 32 + nt * 16 + lrow) * 64;
#pragma unroll
    for (int v = 0; v < 8; ++v) {
      int k0 = ((v & 4) << 3) + (lhalf << 4) + ((v & 3) << 2);
      bfrag[nt][v] = *(const int*)(bp + k0);
    }
  }
#pragma unroll
  for (int mt = 0; mt < 4; ++mt) {
    const char* ap = As + (wm * 64 + mt * 16 + lrow) * 64;
    v8i afrag;
#pragma unroll
    for (int v = 0; v < 8; ++v) {
      int k0 = ((v & 4) << 3) + (lhalf << 4) + ((v & 3) << 2);
      afrag[v] = *(const int*)(ap + k0);
    }
    acc[mt][0] = __builtin_amdgcn_wmma_i32_16x16x64_iu8(
        true, afrag, true, bfrag[0], acc[mt][0], false, false);
    acc[mt][1] = __builtin_amdgcn_wmma_i32_16x16x64_iu8(
        true, afrag, true, bfrag[1], acc[mt][1], false, false);
  }
}

__global__ __launch_bounds__(256, 2) void int8_wmma_gemm_kernel(
    const char* __restrict__ Aq, const char* __restrict__ Bq,
    const float* __restrict__ wscale, const float* __restrict__ iscale,
    const float* __restrict__ bias, float* __restrict__ out,
    int M, int N, int K) {
  __shared__ char As[2][TILE_M * TILE_K];
  __shared__ char Bs[2][TILE_N * TILE_K];

  const int tid   = threadIdx.x;
  const int lane  = tid & 31;
  const int wave  = tid >> 5;
  const int wm    = wave & 1;
  const int wn    = wave >> 1;
  const int lrow  = lane & 15;
  const int lhalf = lane >> 4;

  const int bm = blockIdx.y * TILE_M;
  const int bn = blockIdx.x * TILE_N;

  v8i acc[4][2];
#pragma unroll
  for (int mt = 0; mt < 4; ++mt)
#pragma unroll
    for (int nt = 0; nt < 2; ++nt) acc[mt][nt] = v8i{};

  const int nk = K / TILE_K;

  stage_tiles(Aq, Bq, As[0], Bs[0], bm, bn, 0, K, tid);

#pragma unroll 1
  for (int i = 0; i < nk; i += 2) {
    stage_tiles(Aq, Bq, As[1], Bs[1], bm, bn, (i + 1) * TILE_K, K, tid);
    async_wait_4();
    __syncthreads();
    compute_tile(As[0], Bs[0], acc, wm, wn, lrow, lhalf);
    __syncthreads();

    if (i + 2 < nk) {
      stage_tiles(Aq, Bq, As[0], Bs[0], bm, bn, (i + 2) * TILE_K, K, tid);
      async_wait_4();
    } else {
      async_wait_0();
    }
    __syncthreads();
    compute_tile(As[1], Bs[1], acc, wm, wn, lrow, lhalf);
    __syncthreads();
  }

  const float is0 = iscale[0];
#pragma unroll
  for (int nt = 0; nt < 2; ++nt) {
    int n = bn + wn * 32 + nt * 16 + lrow;
    float sc = wscale[n] * is0;
    float bi = bias[n];
#pragma unroll
    for (int mt = 0; mt < 4; ++mt) {
#pragma unroll
      for (int r = 0; r < 8; ++r) {
        int m = bm + wm * 64 + mt * 16 + (lhalf << 3) + r;
        out[(size_t)m * N + n] = (float)acc[mt][nt][r] * sc + bi;
      }
    }
  }
}

extern "C" void kernel_launch(void* const* d_in, const int* in_sizes, int n_in,
                              void* d_out, int out_size, void* d_ws, size_t ws_size,
                              hipStream_t stream) {
  const float* x  = (const float*)d_in[0];
  const float* qw = (const float*)d_in[1];
  const float* ws = (const float*)d_in[2];
  const float* is = (const float*)d_in[3];
  const float* bs = (const float*)d_in[4];
  float* out = (float*)d_out;

  const int N = in_sizes[2];
  const int K = in_sizes[1] / N;
  const int M = in_sizes[0] / K;

  char* Aq = (char*)d_ws;
  char* Bq = (char*)d_ws + (size_t)M * (size_t)K;

  const int an4 = (M * K) / 4;
  quant_input_i8_kernel<<<(an4 + 255) / 256, 256, 0, stream>>>(
      x, is, (unsigned*)Aq, an4);
  const int wn4 = (N * K) / 4;
  quant_weight_i8_kernel<<<(wn4 + 255) / 256, 256, 0, stream>>>(
      qw, (unsigned*)Bq, wn4);

  dim3 grid(N / TILE_N, M / TILE_M);
  int8_wmma_gemm_kernel<<<grid, 256, 0, stream>>>(
      Aq, Bq, ws, is, bs, out, M, N, K);
}
